// TransLayer_41446434406886
// MI455X (gfx1250) — matrix operand order repaired and checked by reading, not compiled, not run on hardware
//
#include <hip/hip_runtime.h>
#include <hip/hip_bf16.h>

// ---------------- problem constants ----------------
#define DIMX   512
#define HEADS  2
#define DHEAD  64
#define INNERD 128
#define MLAND  256      // landmarks (M)
#define NSEQ   8192
#define BATCH  4
#define BHX    8        // BATCH*HEADS
#define NTOK   32768    // BATCH*NSEQ
#define KERN   33
#define LN_EPS 1e-5f

typedef __attribute__((ext_vector_type(16))) __bf16 v16bf;
typedef __attribute__((ext_vector_type(8)))  __bf16 v8bf;
typedef __attribute__((ext_vector_type(8)))  float  v8f;
typedef __attribute__((ext_vector_type(2)))  float  v2f;

__device__ __forceinline__ v16bf cat8(v8bf lo, v8bf hi) {
  return __builtin_shufflevector(lo, hi, 0, 1, 2, 3, 4, 5, 6, 7,
                                 8, 9, 10, 11, 12, 13, 14, 15);
}

// LDS byte offset of a shared-memory pointer (generic -> AS(3) -> u32)
__device__ __forceinline__ unsigned lds_off_u32(const void* p) {
  return (unsigned)(unsigned long long)(__attribute__((address_space(3))) const char*)p;
}

// CDNA5 async copy global -> LDS (ASYNCcnt tracked), GV addressing
__device__ __forceinline__ void async_ld_b128(unsigned lds_dst, const void* gsrc) {
  asm volatile("global_load_async_to_lds_b128 %0, %1, off"
               :: "v"(lds_dst), "v"((unsigned long long)gsrc) : "memory");
}

// ================= WMMA bf16 16x16 tile core (A @ B^T) =================
// A: row-major M x K (lda);  B: row-major N x K (ldb)  -> C = A @ B^T
// Fragment fetch fully vectorized (2x b128 per operand per k-step):
//   A lane<16: row=idx, K = {hf*8+0..7, hf*8+16..23}
//   B lane:    col=idx, K = {hf*8+0..7, hf*8+16..23}
__device__ __forceinline__ v8f wmma_tile_t(const __bf16* __restrict__ A,
                                           const __bf16* __restrict__ B,
                                           int lda, int ldb, int K,
                                           int row0, int col0) {
  const int lane = threadIdx.x & 31;
  const int hf   = lane >> 4;
  const int idx  = lane & 15;
  const __bf16* arow = A + (size_t)(row0 + idx) * lda + hf * 8;
  const __bf16* brow = B + (size_t)(col0 + idx) * ldb + hf * 8;
  v8f acc = {};
  for (int k0 = 0; k0 < K; k0 += 32) {
    __builtin_prefetch(arow + k0 + 256, 0, 0);   // global_prefetch_b8
    __builtin_prefetch(brow + k0 + 256, 0, 0);
    const v8bf a0 = *(const v8bf*)(arow + k0);
    const v8bf a1 = *(const v8bf*)(arow + k0 + 16);
    const v8bf b0 = *(const v8bf*)(brow + k0);
    const v8bf b1 = *(const v8bf*)(brow + k0 + 16);
    acc = __builtin_amdgcn_wmma_f32_16x16x32_bf16(false, cat8(a0, a1),
                                                  false, cat8(b0, b1),
                                                  (short)0, acc, false, false);
  }
  return acc;
}

// ================= generic batched bf16 GEMM (A @ B^T) =================
// grid: (N/32, M/32, batch), block 128 (4 waves, 2x2 tiles of 16x16)
template<bool TOUT, typename OutT>
__global__ void gemm_t_k(const __bf16* __restrict__ A, const __bf16* __restrict__ B,
                         OutT* __restrict__ C, int Nn, int Kk, int ldc, float scale,
                         size_t sA, size_t sB, size_t sC) {
  const int bh   = blockIdx.z;
  const int wave = threadIdx.x >> 5;
  const int row0 = blockIdx.y * 32 + (wave >> 1) * 16;
  const int col0 = blockIdx.x * 32 + (wave & 1) * 16;
  const v8f acc = wmma_tile_t(A + sA * bh, B + sB * bh, Kk, Kk, Kk, row0, col0);
  const int lane = threadIdx.x & 31, hf = lane >> 4, idx = lane & 15;
  OutT* Cb = C + sC * bh;
  const int col = col0 + idx;
#pragma unroll
  for (int r = 0; r < 8; ++r) {
    const int row = row0 + hf * 8 + r;
    if (TOUT) Cb[(size_t)col * ldc + row] = (OutT)(acc[r] * scale);
    else      Cb[(size_t)row * ldc + col] = (OutT)(acc[r] * scale);
  }
}

// ================= fp32 WMMA 256x256 GEMM (Newton-Schulz) =================
__global__ void gemm256_f32_k(const float* __restrict__ A, const float* __restrict__ B,
                              float* __restrict__ C, float scale) {
  const int bh    = blockIdx.z;
  const float* Ab = A + (size_t)bh * 65536;
  const float* Bb = B + (size_t)bh * 65536;
  float*       Cb = C + (size_t)bh * 65536;
  const int wave = threadIdx.x >> 5;
  const int row0 = blockIdx.y * 32 + (wave >> 1) * 16;
  const int col0 = blockIdx.x * 32 + (wave & 1) * 16;
  const int lane = threadIdx.x & 31, hf = lane >> 4, idx = lane & 15;
  v8f acc = {};
  for (int k0 = 0; k0 < 256; k0 += 4) {
    v2f a, b;
    a.x = Ab[(size_t)(row0 + idx) * 256 + k0 + hf * 2 + 0];
    a.y = Ab[(size_t)(row0 + idx) * 256 + k0 + hf * 2 + 1];
    b.x = Bb[(size_t)(k0 + hf * 2 + 0) * 256 + col0 + idx];
    b.y = Bb[(size_t)(k0 + hf * 2 + 1) * 256 + col0 + idx];
    acc = __builtin_amdgcn_wmma_f32_16x16x4_f32(false, a, false, b,
                                                (short)0, acc, false, false);
  }
#pragma unroll
  for (int r = 0; r < 8; ++r)
    Cb[(size_t)(row0 + hf * 8 + r) * 256 + col0 + idx] = acc[r] * scale;
}

// ========== QKV GEMM: LDS double-buffered via async-to-LDS copies ==========
// C(32768x384) = xn(32768x512) @ wqkvT(384x512)^T ; head-split epilogue.
// Tiles 32x32 bf16 staged with global_load_async_to_lds_b128 (ASYNCcnt),
// fragments read back as ds_load_b128 (rows padded to 80B to stagger banks).
__global__ void qkv_k(const __bf16* __restrict__ xn, const __bf16* __restrict__ wT,
                      __bf16* __restrict__ q, __bf16* __restrict__ k,
                      __bf16* __restrict__ v, __bf16* __restrict__ vT) {
  __shared__ __bf16 tA[2][32][40];
  __shared__ __bf16 tB[2][32][40];
  const int t    = threadIdx.x;
  const int lr   = t >> 2;          // tile row this thread copies
  const int lc   = (t & 3) * 8;     // K-element offset (16B chunks)
  const int row0 = blockIdx.y * 32;
  const int col0 = blockIdx.x * 32;
  const __bf16* gA = xn + (size_t)(row0 + lr) * DIMX + lc;
  const __bf16* gB = wT + (size_t)(col0 + lr) * DIMX + lc;
  const unsigned dA0 = lds_off_u32(&tA[0][lr][lc]), dA1 = lds_off_u32(&tA[1][lr][lc]);
  const unsigned dB0 = lds_off_u32(&tB[0][lr][lc]), dB1 = lds_off_u32(&tB[1][lr][lc]);

  async_ld_b128(dA0, gA);           // stage 0
  async_ld_b128(dB0, gB);

  const int wave = t >> 5, lane = t & 31, hf = lane >> 4, idx = lane & 15;
  const int ar = (wave >> 1) * 16 + idx;   // LDS-local A row
  const int br = (wave & 1) * 16 + idx;    // LDS-local B row
  v8f acc = {};
  for (int k0 = 0, s = 0; k0 < DIMX; k0 += 32, s ^= 1) {
    const bool more = (k0 + 32) < DIMX;
    if (more) {                                // prefetch next stage
      async_ld_b128(s ? dA0 : dA1, gA + k0 + 32);
      async_ld_b128(s ? dB0 : dB1, gB + k0 + 32);
      asm volatile("s_wait_asynccnt 0x2" ::: "memory");
    } else {
      asm volatile("s_wait_asynccnt 0x0" ::: "memory");
    }
    __syncthreads();                           // tile[s] valid for all waves
    const v8bf a0 = *(const v8bf*)&tA[s][ar][hf * 8];
    const v8bf a1 = *(const v8bf*)&tA[s][ar][hf * 8 + 16];
    const v8bf b0 = *(const v8bf*)&tB[s][br][hf * 8];
    const v8bf b1 = *(const v8bf*)&tB[s][br][hf * 8 + 16];
    acc = __builtin_amdgcn_wmma_f32_16x16x32_bf16(false, cat8(a0, a1),
                                                  false, cat8(b0, b1),
                                                  (short)0, acc, false, false);
    __syncthreads();                           // tile[s] free for reuse
  }

  const int col = col0 + (wave & 1) * 16 + idx;
  const int tt = col / INNERD, cc = col % INNERD, hh = cc >> 6, d = cc & 63;
  __bf16* dst = (tt == 0) ? q : (tt == 1) ? k : v;
  const float sc = (tt == 0) ? 0.125f : 1.f;   // DHEAD^-0.5
#pragma unroll
  for (int r = 0; r < 8; ++r) {
    const int row = row0 + (wave >> 1) * 16 + hf * 8 + r;
    const int bb = row >> 13, nn = row & (NSEQ - 1);
    const __bf16 val = (__bf16)(acc[r] * sc);
    dst[(((size_t)(bb * HEADS + hh) * NSEQ + nn) << 6) + d] = val;
    if (tt == 2)  // also keep v^T (64 x 8192 per bh) for the a3@v GEMM
      vT[((size_t)(bb * HEADS + hh) * DHEAD + d) * NSEQ + nn] = val;
  }
}

// ================= elementwise / reduction kernels =================
__global__ void cast_bf16_k(const float* __restrict__ s, __bf16* __restrict__ d, size_t n) {
  size_t id = (size_t)blockIdx.x * blockDim.x + threadIdx.x;
  if (id < n) d[id] = (__bf16)s[id];
}

// d[c*rows + r] = (bf16)s[r*cols + c]
__global__ void castT_bf16_k(const float* __restrict__ s, __bf16* __restrict__ d,
                             int rows, int cols) {
  const size_t id = (size_t)blockIdx.x * 256 + threadIdx.x;
  if (id >= (size_t)rows * cols) return;
  const int r = (int)(id / cols), c = (int)(id % cols);
  d[(size_t)c * rows + r] = (__bf16)s[id];
}

// batched 256x256 transpose-cast: zT[bh][c][r] = z[bh][r][c]
__global__ void ztrans_k(const float* __restrict__ z, __bf16* __restrict__ zT) {
  const size_t id = (size_t)blockIdx.x * 256 + threadIdx.x;
  const size_t bh = id >> 16;
  const int r = (id >> 8) & 255, c = id & 255;
  zT[(bh << 16) + ((size_t)c << 8) + r] = (__bf16)z[id];
}

// LayerNorm over DIMX=512, one block(256) per token row -> bf16
__global__ void ln_k(const float* __restrict__ x, const float* __restrict__ gamma,
                     const float* __restrict__ beta, __bf16* __restrict__ xn) {
  __shared__ float red[256];
  const int t = threadIdx.x;
  const size_t row = blockIdx.x;
  const float* xr = x + row * DIMX;
  float v0 = xr[t], v1 = xr[t + 256];
  red[t] = v0 + v1; __syncthreads();
  for (int o = 128; o; o >>= 1) { if (t < o) red[t] += red[t + o]; __syncthreads(); }
  const float mu = red[0] * (1.f / DIMX); __syncthreads();
  const float d0 = v0 - mu, d1 = v1 - mu;
  red[t] = d0 * d0 + d1 * d1; __syncthreads();
  for (int o = 128; o; o >>= 1) { if (t < o) red[t] += red[t + o]; __syncthreads(); }
  const float inv = rsqrtf(red[0] * (1.f / DIMX) + LN_EPS);
  xn[row * DIMX + t]       = (__bf16)(d0 * inv * gamma[t]       + beta[t]);
  xn[row * DIMX + t + 256] = (__bf16)(d1 * inv * gamma[t + 256] + beta[t + 256]);
}

// landmark means over l=32 blocks -> bf16 (b,h,256,64)
__global__ void landmark_k(const __bf16* __restrict__ q, const __bf16* __restrict__ k,
                           __bf16* __restrict__ qlb, __bf16* __restrict__ klb) {
  const int bh = blockIdx.x >> 8, m = blockIdx.x & 255, d = threadIdx.x;
  const size_t base = (((size_t)bh * NSEQ + m * 32) << 6) + d;
  float sq = 0.f, sk = 0.f;
#pragma unroll 4
  for (int j = 0; j < 32; ++j) { sq += (float)q[base + (j << 6)]; sk += (float)k[base + (j << 6)]; }
  const size_t o = (((size_t)bh * MLAND + m) << 6) + d;
  qlb[o] = (__bf16)(sq * (1.f / 32.f));
  klb[o] = (__bf16)(sk * (1.f / 32.f));
}

// row softmax, one block(256) per row of length L
template<typename OutT>
__global__ void softmax_k(const float* __restrict__ S, OutT* __restrict__ P, int L) {
  __shared__ float red[256];
  const int t = threadIdx.x;
  const float* s = S + (size_t)blockIdx.x * L;
  OutT*        o = P + (size_t)blockIdx.x * L;
  float mx = -3.4e38f;
  for (int i = t; i < L; i += 256) mx = fmaxf(mx, s[i]);
  red[t] = mx; __syncthreads();
  for (int off = 128; off; off >>= 1) { if (t < off) red[t] = fmaxf(red[t], red[t + off]); __syncthreads(); }
  mx = red[0]; __syncthreads();
  float sum = 0.f;
  for (int i = t; i < L; i += 256) sum += __expf(s[i] - mx);
  red[t] = sum; __syncthreads();
  for (int off = 128; off; off >>= 1) { if (t < off) red[t] += red[t + off]; __syncthreads(); }
  const float inv = 1.f / red[0];
  for (int i = t; i < L; i += 256) o[i] = (OutT)(__expf(s[i] - mx) * inv);
}

__global__ void zero2_k(unsigned* s) { if (threadIdx.x < 2) s[threadIdx.x] = 0u; }

// |a2| row/col sums, global max (values >=0 -> uint-bit atomicMax)
__global__ void pinv_scale_k(const float* __restrict__ a2, unsigned* __restrict__ scal) {
  __shared__ float rc[256], rr[256];
  const int bh = blockIdx.x, i = threadIdx.x;
  const float* A = a2 + (size_t)bh * 65536;
  float cs = 0.f, rs = 0.f;
  for (int j = 0; j < 256; ++j) { cs += fabsf(A[i * 256 + j]); rs += fabsf(A[j * 256 + i]); }
  rc[i] = cs; rr[i] = rs; __syncthreads();
  for (int o = 128; o; o >>= 1) {
    if (i < o) { rc[i] = fmaxf(rc[i], rc[i + o]); rr[i] = fmaxf(rr[i], rr[i + o]); }
    __syncthreads();
  }
  if (i == 0) { atomicMax(&scal[0], __float_as_uint(rc[0])); atomicMax(&scal[1], __float_as_uint(rr[0])); }
}

__global__ void zinit_k(const float* __restrict__ a2, const unsigned* __restrict__ scal,
                        float* __restrict__ z) {
  const size_t id = (size_t)blockIdx.x * 256 + threadIdx.x;
  const float denom = __uint_as_float(scal[0]) * __uint_as_float(scal[1]);
  const int c = id & 255, r = (id >> 8) & 255;
  const size_t bh = id >> 16;
  z[id] = a2[(bh << 16) + ((size_t)c << 8) + r] / denom;
}

// O = c*I - X  (256x256 batched, flat)
__global__ void diag_sub_k(const float* __restrict__ X, float* __restrict__ O, float c) {
  const size_t id = (size_t)blockIdx.x * 256 + threadIdx.x;
  const int r = (id >> 8) & 255, cc = id & 255;
  O[id] = ((r == cc) ? c : 0.f) - X[id];
}

// depthwise conv over sequence (kernel 33, zero pad 16), per head weights
__global__ void conv_k(const __bf16* __restrict__ v, const float* __restrict__ w_res,
                       float* __restrict__ res) {
  const size_t id = (size_t)blockIdx.x * 256 + threadIdx.x; // BH*NSEQ*64
  const int d = id & 63;
  const size_t tmp = id >> 6;
  const int nn = (int)(tmp & (NSEQ - 1));
  const int bh = (int)(tmp >> 13);
  const int h = bh & 1;
  float s = 0.f;
#pragma unroll
  for (int kk = 0; kk < KERN; ++kk) {
    const int p = nn + kk - (KERN / 2);
    if (p >= 0 && p < NSEQ)
      s += w_res[h * KERN + kk] * (float)v[(((size_t)bh * NSEQ + p) << 6) + d];
  }
  res[id] = s;
}

// attn = w1(8192x256) @ a3vT(64x256)^T + conv-res -> outh (b, n, h*64+d) bf16
__global__ void attn_k(const __bf16* __restrict__ w1, const __bf16* __restrict__ a3vT,
                       const float* __restrict__ res, __bf16* __restrict__ outh) {
  const int bh   = blockIdx.z;
  const int wave = threadIdx.x >> 5;
  const int row0 = blockIdx.y * 32 + (wave >> 1) * 16;
  const int col0 = blockIdx.x * 32 + (wave & 1) * 16;
  const v8f acc = wmma_tile_t(w1 + (size_t)bh * NSEQ * MLAND,
                              a3vT + (size_t)bh * DHEAD * MLAND,
                              MLAND, MLAND, MLAND, row0, col0);
  const int lane = threadIdx.x & 31, hf = lane >> 4, idx = lane & 15;
  const int d = col0 + idx, b = bh >> 1, h = bh & 1;
#pragma unroll
  for (int r = 0; r < 8; ++r) {
    const int nn = row0 + hf * 8 + r;
    const float val = acc[r] + res[(((size_t)bh * NSEQ + nn) << 6) + d];
    outh[((size_t)(b * NSEQ + nn)) * INNERD + h * DHEAD + d] = (__bf16)val;
  }
}

// y = x + outh @ woutT(512x128)^T + b_out  -> fp32 d_out
__global__ void out_k(const __bf16* __restrict__ outh, const __bf16* __restrict__ woutT,
                      const float* __restrict__ bout, const float* __restrict__ x,
                      float* __restrict__ y) {
  const int wave = threadIdx.x >> 5;
  const int row0 = blockIdx.y * 32 + (wave >> 1) * 16;
  const int col0 = blockIdx.x * 32 + (wave & 1) * 16;
  const v8f acc = wmma_tile_t(outh, woutT, INNERD, INNERD, INNERD, row0, col0);
  const int lane = threadIdx.x & 31, hf = lane >> 4, idx = lane & 15;
  const int col = col0 + idx;
  const float bo = bout[col];
#pragma unroll
  for (int r = 0; r < 8; ++r) {
    const size_t o = (size_t)(row0 + hf * 8 + r) * DIMX + col;
    y[o] = x[o] + acc[r] + bo;
  }
}

// ================= host orchestration =================
extern "C" void kernel_launch(void* const* d_in, const int* in_sizes, int n_in,
                              void* d_out, int out_size, void* d_ws, size_t ws_size,
                              hipStream_t stream) {
  const float* x     = (const float*)d_in[0];
  const float* gamma = (const float*)d_in[1];
  const float* beta  = (const float*)d_in[2];
  const float* w_qkv = (const float*)d_in[3];
  const float* w_out = (const float*)d_in[4];
  const float* b_out = (const float*)d_in[5];
  const float* w_res = (const float*)d_in[6];
  float* y = (float*)d_out;

  // -------- workspace carve-out --------
  char* p = (char*)d_ws;
  auto alloc = [&](size_t bytes) -> void* {
    void* r = (void*)p; p += (bytes + 255) & ~(size_t)255; return r;
  };
  __bf16* xn     = (__bf16*)alloc((size_t)NTOK * DIMX * 2);
  __bf16* wqkvT  = (__bf16*)alloc((size_t)DIMX * 3 * INNERD * 2);  // (384 x 512)
  __bf16* woutT  = (__bf16*)alloc((size_t)INNERD * DIMX * 2);      // (512 x 128)
  __bf16* qb     = (__bf16*)alloc((size_t)BHX * NSEQ * DHEAD * 2);
  __bf16* kb     = (__bf16*)alloc((size_t)BHX * NSEQ * DHEAD * 2);
  __bf16* vb     = (__bf16*)alloc((size_t)BHX * NSEQ * DHEAD * 2);
  __bf16* vT     = (__bf16*)alloc((size_t)BHX * DHEAD * NSEQ * 2); // (64 x 8192)/bh
  __bf16* qlb    = (__bf16*)alloc((size_t)BHX * MLAND * DHEAD * 2);
  __bf16* klb    = (__bf16*)alloc((size_t)BHX * MLAND * DHEAD * 2);
  float*  simf   = (float*) alloc((size_t)BHX * NSEQ * MLAND * 4); // reused sim1/sim3
  __bf16* a1b    = (__bf16*)alloc((size_t)BHX * NSEQ * MLAND * 2);
  __bf16* a3b    = (__bf16*)alloc((size_t)BHX * MLAND * NSEQ * 2);
  float*  a2f    = (float*) alloc((size_t)BHX * 65536 * 4);
  float*  zf     = (float*) alloc((size_t)BHX * 65536 * 4);
  float*  znf    = (float*) alloc((size_t)BHX * 65536 * 4);
  float*  xzf    = (float*) alloc((size_t)BHX * 65536 * 4);
  float*  taf    = (float*) alloc((size_t)BHX * 65536 * 4);
  float*  tbf    = (float*) alloc((size_t)BHX * 65536 * 4);
  __bf16* zTb    = (__bf16*)alloc((size_t)BHX * 65536 * 2);        // z^T per bh
  __bf16* a3vTb  = (__bf16*)alloc((size_t)BHX * DHEAD * MLAND * 2);// (64 x 256)/bh
  __bf16* w1b    = (__bf16*)alloc((size_t)BHX * NSEQ * MLAND * 2);
  float*  resf   = (float*) alloc((size_t)BHX * NSEQ * DHEAD * 4);
  __bf16* outhb  = (__bf16*)alloc((size_t)NTOK * INNERD * 2);
  unsigned* scal = (unsigned*)alloc(64);

  // -------- transposed weight casts + layernorm --------
  castT_bf16_k<<<(DIMX * 3 * INNERD) / 256, 256, 0, stream>>>(w_qkv, wqkvT, DIMX, 3 * INNERD);
  castT_bf16_k<<<(INNERD * DIMX) / 256, 256, 0, stream>>>(w_out, woutT, INNERD, DIMX);
  ln_k<<<NTOK, 256, 0, stream>>>(x, gamma, beta, xn);

  // -------- QKV projection (async-LDS staged WMMA) --------
  qkv_k<<<dim3((3 * INNERD) / 32, NTOK / 32, 1), 128, 0, stream>>>(xn, wqkvT, qb, kb, vb, vT);

  // -------- landmarks --------
  landmark_k<<<BHX * MLAND, DHEAD, 0, stream>>>(qb, kb, qlb, klb);

  // -------- sim1 = q @ k_l^T ; a1 = softmax --------
  gemm_t_k<false, float><<<dim3(MLAND / 32, NSEQ / 32, BHX), 128, 0, stream>>>(
      qb, klb, simf, MLAND, DHEAD, MLAND, 1.f,
      (size_t)NSEQ * DHEAD, (size_t)MLAND * DHEAD, (size_t)NSEQ * MLAND);
  softmax_k<__bf16><<<BHX * NSEQ, 256, 0, stream>>>(simf, a1b, MLAND);

  // -------- sim2 = q_l @ k_l^T ; a2 = softmax (fp32) --------
  gemm_t_k<false, float><<<dim3(MLAND / 32, MLAND / 32, BHX), 128, 0, stream>>>(
      qlb, klb, xzf, MLAND, DHEAD, MLAND, 1.f,
      (size_t)MLAND * DHEAD, (size_t)MLAND * DHEAD, (size_t)MLAND * MLAND);
  softmax_k<float><<<BHX * MLAND, 256, 0, stream>>>(xzf, a2f, MLAND);

  // -------- sim3 = q_l @ k^T ; a3 = softmax over 8192 --------
  gemm_t_k<false, float><<<dim3(NSEQ / 32, MLAND / 32, BHX), 128, 0, stream>>>(
      qlb, kb, simf, NSEQ, DHEAD, NSEQ, 1.f,
      (size_t)MLAND * DHEAD, (size_t)NSEQ * DHEAD, (size_t)MLAND * NSEQ);
  softmax_k<__bf16><<<BHX * MLAND, 256, 0, stream>>>(simf, a3b, NSEQ);

  // -------- Moore-Penrose pinv of a2 (fp32 WMMA) --------
  zero2_k<<<1, 32, 0, stream>>>(scal);
  pinv_scale_k<<<BHX, 256, 0, stream>>>(a2f, scal);
  zinit_k<<<(BHX * 65536) / 256, 256, 0, stream>>>(a2f, scal, zf);
  const dim3 g256(8, 8, BHX);
  const int  ew = (BHX * 65536) / 256;
  float* zc = zf; float* zn = znf;
  for (int it = 0; it < 6; ++it) {
    gemm256_f32_k<<<g256, 128, 0, stream>>>(a2f, zc, xzf, 1.f);   // XZ = a2 @ z
    diag_sub_k<<<ew, 256, 0, stream>>>(xzf, taf, 7.f);            // TA = 7I - XZ
    gemm256_f32_k<<<g256, 128, 0, stream>>>(xzf, taf, tbf, 1.f);  // TB = XZ @ TA
    diag_sub_k<<<ew, 256, 0, stream>>>(tbf, taf, 15.f);           // TA = 15I - TB
    gemm256_f32_k<<<g256, 128, 0, stream>>>(xzf, taf, tbf, 1.f);  // TB = XZ @ TA
    diag_sub_k<<<ew, 256, 0, stream>>>(tbf, taf, 13.f);           // TA = 13I - TB
    gemm256_f32_k<<<g256, 128, 0, stream>>>(zc, taf, zn, 0.25f);  // z' = 0.25 z @ TA
    float* t = zc; zc = zn; zn = t;
  }
  ztrans_k<<<(BHX * 65536) / 256, 256, 0, stream>>>(zc, zTb);

  // -------- a3v^T = (a3 @ v)^T via TOUT (K=8192, B = v^T) --------
  gemm_t_k<true, __bf16><<<dim3(DHEAD / 32, MLAND / 32, BHX), 128, 0, stream>>>(
      a3b, vT, a3vTb, DHEAD, NSEQ, MLAND, 1.f,
      (size_t)MLAND * NSEQ, (size_t)DHEAD * NSEQ, (size_t)DHEAD * MLAND);

  // -------- w1 = a1 @ a2_inv  (B = z^T) --------
  gemm_t_k<false, __bf16><<<dim3(MLAND / 32, NSEQ / 32, BHX), 128, 0, stream>>>(
      a1b, zTb, w1b, MLAND, MLAND, MLAND, 1.f,
      (size_t)NSEQ * MLAND, (size_t)MLAND * MLAND, (size_t)NSEQ * MLAND);

  // -------- depthwise conv residual on v --------
  conv_k<<<(BHX * NSEQ * DHEAD) / 256, 256, 0, stream>>>(vb, w_res, resf);

  // -------- attn = w1 @ a3v + res -> head-concat bf16 --------
  attn_k<<<dim3(DHEAD / 32, NSEQ / 32, BHX), 128, 0, stream>>>(w1b, a3vTb, resf, outhb);

  // -------- y = x + outh @ w_out + b_out --------
  out_k<<<dim3(DIMX / 32, NTOK / 32, 1), 128, 0, stream>>>(outhb, woutT, b_out, x, y);
}
